// CSWinSelfAttention_45569603010908
// MI455X (gfx1250) — compile-verified, matrix operand order checked by reading, not run on hardware
//
#include <hip/hip_runtime.h>
#include <hip/hip_bf16.h>

// ---------------------------------------------------------------------------
// CSWin self-attention block for MI455X (gfx1250, wave32, WMMA).
// All matmuls via v_wmma_f32_16x16x32_bf16 (bf16 in, fp32 accum).
// Softmax / LayerNorm statistics in fp32 with 16-lane (softmax) / 32-lane (LayerNorm) butterfly reductions.
// Intermediates in bf16 to halve HBM traffic (23.3 TB/s roofline bound).
// K-tile staging uses gfx1250 async global->LDS copies (ASYNCcnt path).
// Workspace use: xn(64MB) + qkv(192MB) + h(64MB) + weightsT (<1MB) ~= 336MB.
// ---------------------------------------------------------------------------

#define DIM    256
#define QKV_N  768
#define NTOK   (8 * 128 * 128)   // 131072 tokens
#define HD     32                // head dim
#define SEQ    256               // stripe sequence length (2*128)

typedef __bf16 bf16_t;
typedef __attribute__((ext_vector_type(8)))  __bf16 v8bf;
typedef __attribute__((ext_vector_type(16))) __bf16 v16bf;
typedef __attribute__((ext_vector_type(8)))  float  v8f;

union BF16Frag { v16bf v; v8bf h[2]; };

__device__ __forceinline__ v8f wmma_bf16(v16bf a, v16bf b, v8f c) {
  // (neg_a, A, neg_b, B, c_mod, C, reuse_a, reuse_b)
  return __builtin_amdgcn_wmma_f32_16x16x32_bf16(false, a, false, b, (short)0, c,
                                                 false, false);
}

// Async global->LDS copy, 16 bytes per lane. VDST VGPR holds the LDS byte
// address (generic LDS pointers keep the LDS offset in the low 32 bits per
// the aperture rules); VADDR holds the 64-bit global address. ASYNCcnt.
__device__ __forceinline__ void async_ld_b128(unsigned lds_addr,
                                              const bf16_t* gptr) {
  asm volatile("global_load_async_to_lds_b128 %0, %1, off"
               :: "v"(lds_addr), "v"(gptr)
               : "memory");
}

__device__ __forceinline__ void wait_asynccnt0() {
#if __has_builtin(__builtin_amdgcn_s_wait_asynccnt)
  __builtin_amdgcn_s_wait_asynccnt(0);
#else
  asm volatile("s_wait_asynccnt 0" ::: "memory");
#endif
}

// Load a 16x32 bf16 fragment (A-matrix layout; also used for B^T rows) from a
// row-major buffer p with leading dimension ld (elements).
// Per ISA: lanes 0-15 hold row (l&15), K chunks [0,8)+[16,24);
//          lanes 16-31 hold same rows, K chunks [8,16)+[24,32).
__device__ __forceinline__ v16bf ld_frag(const bf16_t* p, int ld) {
  const int l   = threadIdx.x & 31;
  const int row = l & 15;
  const int kh  = (l >> 4) << 3;
  BF16Frag u;
  u.h[0] = *(const v8bf*)(p + (size_t)row * ld + kh);
  u.h[1] = *(const v8bf*)(p + (size_t)row * ld + kh + 16);
  return u.v;
}

// 16-lane butterfly reductions (reduce across n-columns of a C-layout tile;
// masks < 16 keep the two EXEC halves independent, matching C-layout halves).
__device__ __forceinline__ float halfmax(float v) {
#pragma unroll
  for (int m = 1; m < 16; m <<= 1) v = fmaxf(v, __shfl_xor(v, m, 32));
  return v;
}
__device__ __forceinline__ float halfsum(float v) {
#pragma unroll
  for (int m = 1; m < 16; m <<= 1) v += __shfl_xor(v, m, 32);
  return v;
}

// Token index of sequence row r inside stripe unit (dir, b, s).
// dir==0 (horizontal): t = b*16384 + s*256 + r          (contiguous rows!)
// dir==1 (vertical):   t = b*16384 + (r&127)*128 + s*2 + (r>>7)
__device__ __forceinline__ int token_of(int dir, int b, int s, int r) {
  return dir == 0 ? (b * 16384 + s * 256 + r)
                  : (b * 16384 + (r & 127) * 128 + s * 2 + (r >> 7));
}

// ---------------------------------------------------------------------------
// Weight prep: transpose fp32 (K,N) -> bf16 (N,K) so B-fragments are two
// contiguous 16B loads per lane.
// ---------------------------------------------------------------------------
__global__ __launch_bounds__(256) void prep_weights_kernel(
    const float* __restrict__ Wqkv, const float* __restrict__ Wproj,
    bf16_t* __restrict__ WqkvT, bf16_t* __restrict__ WprojT) {
  int idx = blockIdx.x * 256 + threadIdx.x;
  if (idx < QKV_N * DIM) {
    int n = idx / DIM, k = idx % DIM;
    WqkvT[idx] = (bf16_t)Wqkv[(size_t)k * QKV_N + n];
  } else {
    int j = idx - QKV_N * DIM;
    if (j < DIM * DIM) {
      int n = j / DIM, k = j % DIM;
      WprojT[j] = (bf16_t)Wproj[(size_t)k * DIM + n];
    }
  }
}

// ---------------------------------------------------------------------------
// LayerNorm: one wave per token (8 ch/lane), fp32 stats, bf16 output.
// ---------------------------------------------------------------------------
__global__ __launch_bounds__(256) void ln_kernel(
    const float* __restrict__ x, const float* __restrict__ gamma,
    const float* __restrict__ beta, bf16_t* __restrict__ xn) {
  const int wid = threadIdx.x >> 5, l = threadIdx.x & 31;
  const size_t tok = (size_t)blockIdx.x * 8 + wid;
  const float* xp = x + tok * DIM;
  float v[8];
  float sum = 0.f;
#pragma unroll
  for (int i = 0; i < 8; i++) { v[i] = xp[l + 32 * i]; sum += v[i]; }
#pragma unroll
  for (int m = 1; m < 32; m <<= 1) sum += __shfl_xor(sum, m, 32);
  const float mean = sum * (1.0f / DIM);
  float sq = 0.f;
#pragma unroll
  for (int i = 0; i < 8; i++) { float d = v[i] - mean; sq += d * d; }
#pragma unroll
  for (int m = 1; m < 32; m <<= 1) sq += __shfl_xor(sq, m, 32);
  const float rstd = rsqrtf(sq * (1.0f / DIM) + 1e-5f);
  bf16_t* op = xn + tok * DIM;
#pragma unroll
  for (int i = 0; i < 8; i++) {
    int c = l + 32 * i;
    op[c] = (bf16_t)((v[i] - mean) * rstd * gamma[c] + beta[c]);
  }
}

// ---------------------------------------------------------------------------
// QKV GEMM: xn(NTOK x 256) @ Wqkv(256 x 768) + bqkv -> qkv bf16.
// Block = 256 thr (8 waves) = 64(M) x 256(N) tile; wave = 32x64; K-loop 8
// WMMA steps. Weights (<=384KB bf16) live in L2 across all M-blocks.
// ---------------------------------------------------------------------------
__global__ __launch_bounds__(256) void qkv_gemm_kernel(
    const bf16_t* __restrict__ xn, const bf16_t* __restrict__ WT,
    const float* __restrict__ bias, bf16_t* __restrict__ qkv) {
  const int wid = threadIdx.x >> 5;
  const int mw = wid >> 2, nw = wid & 3;
  const int row0 = blockIdx.y * 64 + mw * 32;
  const int col0 = blockIdx.x * 256 + nw * 64;
  v8f acc[2][4] = {};
  for (int k0 = 0; k0 < DIM; k0 += 32) {
    __builtin_prefetch(xn + (size_t)(row0 + 64) * DIM + k0, 0, 1);
    v16bf a0 = ld_frag(xn + (size_t)row0 * DIM + k0, DIM);
    v16bf a1 = ld_frag(xn + (size_t)(row0 + 16) * DIM + k0, DIM);
#pragma unroll
    for (int nt = 0; nt < 4; nt++) {
      v16bf bfr = ld_frag(WT + (size_t)(col0 + nt * 16) * DIM + k0, DIM);
      acc[0][nt] = wmma_bf16(a0, bfr, acc[0][nt]);
      acc[1][nt] = wmma_bf16(a1, bfr, acc[1][nt]);
    }
  }
  const int l = threadIdx.x & 31;
  const int nn = l & 15, half8 = (l >> 4) << 3;
#pragma unroll
  for (int mf = 0; mf < 2; mf++)
#pragma unroll
    for (int nt = 0; nt < 4; nt++)
#pragma unroll
      for (int r = 0; r < 8; r++) {
        int row = row0 + mf * 16 + half8 + r;
        int col = col0 + nt * 16 + nn;
        qkv[(size_t)row * QKV_N + col] = (bf16_t)(acc[mf][nt][r] + bias[col]);
      }
}

// ---------------------------------------------------------------------------
// Stripe attention: one block per (dir, unit, head) -> 2*512*4 = 4096 blocks.
// seq=256, hd=32 (== WMMA K). K staged via async global->LDS (ASYNCcnt);
// V staged transposed through VGPRs; two-pass softmax; P routed through
// wave-private LDS to convert C-layout -> A-fragment layout.
// ---------------------------------------------------------------------------
__global__ __launch_bounds__(256) void attn_kernel(
    const bf16_t* __restrict__ qkv, bf16_t* __restrict__ hbuf) {
  __shared__ __align__(16) bf16_t Ksh[SEQ * HD];      // 16KB keys row-major
  __shared__ __align__(16) bf16_t Vt[HD * SEQ];       // 16KB V transposed
  __shared__ __align__(16) bf16_t Psh[8 * 32 * 32];   // 16KB wave-private P

  const int bx   = blockIdx.x;
  const int dir  = bx >> 11;
  const int rem  = bx & 2047;
  const int u    = rem >> 2;
  const int head = rem & 3;
  const int b    = u >> 6;
  const int s    = u & 63;
  const int choff = dir * 128 + head * 32;   // q / out channel base
  const int koff  = DIM + choff;
  const int voff  = 2 * DIM + choff;

  const int wid = threadIdx.x >> 5;
  const int m0  = wid * 32;                 // this wave's 32 query rows
  const int l   = threadIdx.x & 31;
  const int half8 = (l >> 4) << 3;
  const int nn    = l & 15;
  const float scale = 0.17677669529663687f; // 32^-0.5

  // Kick off async K staging first: 4x16B per thread, no VGPR round-trip.
  {
    const int key = threadIdx.x;
    const bf16_t* kp =
        qkv + (size_t)token_of(dir, b, s, key) * QKV_N + koff;
    const unsigned kl = (unsigned)(size_t)(const void*)&Ksh[key * HD];
#pragma unroll
    for (int c = 0; c < 4; c++) async_ld_b128(kl + 16 * c, kp + 8 * c);
  }

  // V staged transposed (async copies can't transpose): VGPR round-trip.
  {
    const int key = threadIdx.x;
    const bf16_t* vp =
        qkv + (size_t)token_of(dir, b, s, key) * QKV_N + voff;
    bf16_t vv[32];
#pragma unroll
    for (int c = 0; c < 4; c++)
      *(v8bf*)(&vv[8 * c]) = *(const v8bf*)(vp + 8 * c);
#pragma unroll
    for (int c = 0; c < 32; c++) Vt[c * SEQ + key] = vv[c];
  }

  // Gathered Q fragments (two 16-row fragments, K = hd = 32); overlaps the
  // in-flight async K copy.
  v16bf qa0, qa1;
  {
    BF16Frag u0, u1;
    const int kh = (l >> 4) << 3;
    const bf16_t* p0 =
        qkv + (size_t)token_of(dir, b, s, m0 + (l & 15)) * QKV_N + choff;
    const bf16_t* p1 =
        qkv + (size_t)token_of(dir, b, s, m0 + 16 + (l & 15)) * QKV_N + choff;
    u0.h[0] = *(const v8bf*)(p0 + kh);  u0.h[1] = *(const v8bf*)(p0 + kh + 16);
    u1.h[0] = *(const v8bf*)(p1 + kh);  u1.h[1] = *(const v8bf*)(p1 + kh + 16);
    qa0 = u0.v; qa1 = u1.v;
  }

  wait_asynccnt0();   // K copy landed in LDS
  __syncthreads();

  // ---- Pass 1: row maxima (per-lane running max, one butterfly at end) ----
  float rm0[8], rm1[8];
#pragma unroll
  for (int r = 0; r < 8; r++) { rm0[r] = -1e30f; rm1[r] = -1e30f; }
  for (int kt = 0; kt < 16; kt++) {
    v16bf kb = ld_frag(&Ksh[kt * 16 * HD], HD);
    v8f z = {};
    v8f s0 = wmma_bf16(qa0, kb, z);
    v8f s1 = wmma_bf16(qa1, kb, z);
#pragma unroll
    for (int r = 0; r < 8; r++) {
      rm0[r] = fmaxf(rm0[r], s0[r] * scale);
      rm1[r] = fmaxf(rm1[r], s1[r] * scale);
    }
  }
#pragma unroll
  for (int r = 0; r < 8; r++) { rm0[r] = halfmax(rm0[r]); rm1[r] = halfmax(rm1[r]); }

  // ---- Pass 2: exp, row sums, O = P @ V accumulated over key pairs -------
  float rs0[8] = {}, rs1[8] = {};
  v8f o00 = {}, o01 = {}, o10 = {}, o11 = {};
  bf16_t* pw = &Psh[wid * 32 * 32];          // wave-private 32x32 P staging
  for (int ktp = 0; ktp < 8; ktp++) {
#pragma unroll
    for (int sub = 0; sub < 2; sub++) {
      const int kt = ktp * 2 + sub;
      v16bf kb = ld_frag(&Ksh[kt * 16 * HD], HD);
      v8f z = {};
      v8f s0 = wmma_bf16(qa0, kb, z);
      v8f s1 = wmma_bf16(qa1, kb, z);
#pragma unroll
      for (int r = 0; r < 8; r++) {
        float p0 = __expf(s0[r] * scale - rm0[r]);
        float p1 = __expf(s1[r] * scale - rm1[r]);
        rs0[r] += p0; rs1[r] += p1;
        // C-layout (m = r+8*half, n = lane&15) -> row-major P staging
        pw[(half8 + r) * 32 + sub * 16 + nn]      = (bf16_t)p0;
        pw[(16 + half8 + r) * 32 + sub * 16 + nn] = (bf16_t)p1;
      }
    }
    // Same-wave LDS store->load kept in order by DS pipeline.
    v16bf pa0 = ld_frag(pw, 32);
    v16bf pa1 = ld_frag(pw + 16 * 32, 32);
    v16bf vb0 = ld_frag(&Vt[0 * SEQ + ktp * 32], SEQ);
    v16bf vb1 = ld_frag(&Vt[16 * SEQ + ktp * 32], SEQ);
    o00 = wmma_bf16(pa0, vb0, o00);
    o01 = wmma_bf16(pa0, vb1, o01);
    o10 = wmma_bf16(pa1, vb0, o10);
    o11 = wmma_bf16(pa1, vb1, o11);
  }
#pragma unroll
  for (int r = 0; r < 8; r++) { rs0[r] = halfsum(rs0[r]); rs1[r] = halfsum(rs1[r]); }

  // Normalize and scatter to h (bf16). Lane mapping of O matches rs exactly.
#pragma unroll
  for (int r = 0; r < 8; r++) {
    const float i0 = 1.0f / rs0[r], i1 = 1.0f / rs1[r];
    const int row0r = m0 + half8 + r;
    const int row1r = m0 + 16 + half8 + r;
    const size_t t0 = (size_t)token_of(dir, b, s, row0r) * DIM + choff;
    const size_t t1 = (size_t)token_of(dir, b, s, row1r) * DIM + choff;
    hbuf[t0 + nn]      = (bf16_t)(o00[r] * i0);
    hbuf[t0 + 16 + nn] = (bf16_t)(o01[r] * i0);
    hbuf[t1 + nn]      = (bf16_t)(o10[r] * i1);
    hbuf[t1 + 16 + nn] = (bf16_t)(o11[r] * i1);
  }
}

// ---------------------------------------------------------------------------
// Output projection: h(NTOK x 256) @ Wproj(256 x 256) + bproj + x -> fp32 out.
// ---------------------------------------------------------------------------
__global__ __launch_bounds__(256) void proj_gemm_kernel(
    const bf16_t* __restrict__ hbuf, const bf16_t* __restrict__ WT,
    const float* __restrict__ bias, const float* __restrict__ x,
    float* __restrict__ out) {
  const int wid = threadIdx.x >> 5;
  const int mw = wid >> 2, nw = wid & 3;
  const int row0 = blockIdx.y * 64 + mw * 32;
  const int col0 = nw * 64;                  // N == 256 == one block tile
  v8f acc[2][4] = {};
  for (int k0 = 0; k0 < DIM; k0 += 32) {
    __builtin_prefetch(hbuf + (size_t)(row0 + 64) * DIM + k0, 0, 1);
    v16bf a0 = ld_frag(hbuf + (size_t)row0 * DIM + k0, DIM);
    v16bf a1 = ld_frag(hbuf + (size_t)(row0 + 16) * DIM + k0, DIM);
#pragma unroll
    for (int nt = 0; nt < 4; nt++) {
      v16bf bfr = ld_frag(WT + (size_t)(col0 + nt * 16) * DIM + k0, DIM);
      acc[0][nt] = wmma_bf16(a0, bfr, acc[0][nt]);
      acc[1][nt] = wmma_bf16(a1, bfr, acc[1][nt]);
    }
  }
  const int l = threadIdx.x & 31;
  const int nn = l & 15, half8 = (l >> 4) << 3;
#pragma unroll
  for (int mf = 0; mf < 2; mf++)
#pragma unroll
    for (int nt = 0; nt < 4; nt++)
#pragma unroll
      for (int r = 0; r < 8; r++) {
        int row = row0 + mf * 16 + half8 + r;
        int col = col0 + nt * 16 + nn;
        size_t idx = (size_t)row * DIM + col;
        out[idx] = acc[mf][nt][r] + bias[col] + x[idx];
      }
}

// ---------------------------------------------------------------------------
extern "C" void kernel_launch(void* const* d_in, const int* in_sizes, int n_in,
                              void* d_out, int out_size, void* d_ws,
                              size_t ws_size, hipStream_t stream) {
  const float* x     = (const float*)d_in[0];
  const float* Wqkv  = (const float*)d_in[1];
  const float* bqkv  = (const float*)d_in[2];
  const float* Wproj = (const float*)d_in[3];
  const float* bproj = (const float*)d_in[4];
  const float* gamma = (const float*)d_in[5];
  const float* beta  = (const float*)d_in[6];
  float* out = (float*)d_out;

  // Workspace layout (bf16): xn | qkv | h | WqkvT | WprojT  (~336 MB)
  bf16_t* xn     = (bf16_t*)d_ws;
  bf16_t* qkv    = xn + (size_t)NTOK * DIM;
  bf16_t* hbuf   = qkv + (size_t)NTOK * QKV_N;
  bf16_t* WqkvT  = hbuf + (size_t)NTOK * DIM;
  bf16_t* WprojT = WqkvT + (size_t)QKV_N * DIM;

  prep_weights_kernel<<<(QKV_N * DIM + DIM * DIM + 255) / 256, 256, 0, stream>>>(
      Wqkv, Wproj, WqkvT, WprojT);
  ln_kernel<<<NTOK / 8, 256, 0, stream>>>(x, gamma, beta, xn);
  qkv_gemm_kernel<<<dim3(QKV_N / 256, NTOK / 64), 256, 0, stream>>>(
      xn, WqkvT, bqkv, qkv);
  attn_kernel<<<4096, 256, 0, stream>>>(qkv, hbuf);
  proj_gemm_kernel<<<dim3(1, NTOK / 64), 256, 0, stream>>>(
      hbuf, WprojT, bproj, x, out);
}
